// SoftAttentionAggregator_71588514889788
// MI455X (gfx1250) — compile-verified, matrix operand order checked by reading, not run on hardware
//
#include <hip/hip_runtime.h>
#include <hip/hip_bf16.h>

// Problem constants (from reference): B=8, H=256, P=1024, K=32, E=4, BN=64
#define BB 8
#define HH 256
#define PP1 1024
#define KK 32
#define EE 4

typedef __bf16 bf16;
typedef __attribute__((ext_vector_type(16))) __bf16 v16bf;
typedef __attribute__((ext_vector_type(8)))  __bf16 v8bf;
typedef __attribute__((ext_vector_type(8)))  float  v8f;

// Staged row pitch in bf16 elements: 296 = 37 * 8 -> 592B = 37 * 16B per row
// (odd number of 16B granules => conflict-free strided b128 row access)
#define PITCH 296
#define ROWB  (PITCH * 2)

// LDS layout (bytes)
#define OFF_A1   0                          // bf16 [64][PITCH]: w1n (c<256) | w1e (256..259) | zeros (..287)
#define OFF_A2   (OFF_A1 + 64 * ROWB)       // bf16 [64][PITCH]: w1c
#define OFF_CURT (OFF_A2 + 64 * ROWB)       // bf16 [32][PITCH]: current^T (rows = p-local, cols = c)
#define OFF_CURB (OFF_CURT + 32 * ROWB)     // f32  [64][33]   : (w1c@current + b1)[o][p-local]
#define OFF_NF   (OFF_CURB + 64 * 33 * 4)   // bf16 8 x [32][PITCH]: per-wave NF^T tile (rows=k, cols=c|edge|0)
#define OFF_WT   (OFF_NF + 8 * 32 * ROWB)   // f32  [8][32]    : per-wave softmax weights
#define OFF_POOL (OFF_WT + 8 * 32 * 4)      // f32  [256][33]  : pooled output staging
#define SMEM_TOTAL (OFF_POOL + 256 * 33 * 4) // = 289,536 bytes < 320 KB

__device__ __forceinline__ v8f wmma_bf16(v16bf a, v16bf b, v8f c) {
  // D = A(16x32 bf16) * B(32x16 bf16) + C(16x16 f32)
  return __builtin_amdgcn_wmma_f32_16x16x32_bf16(false, a, false, b, (short)0, c,
                                                 false, false);
}

// wave32 lane-xor reductions via ds_swizzle (group-of-32: imm = xor<<10 | and=0x1f)
template <int M>
__device__ __forceinline__ float xor_add(float x) {
  int y = __builtin_amdgcn_ds_swizzle(__float_as_int(x), (M << 10) | 0x1F);
  return x + __int_as_float(y);
}
template <int M>
__device__ __forceinline__ float xor_max(float x) {
  int y = __builtin_amdgcn_ds_swizzle(__float_as_int(x), (M << 10) | 0x1F);
  return fmaxf(x, __int_as_float(y));
}
__device__ __forceinline__ float xor16_pair_add(float x) {
  int y = __builtin_amdgcn_ds_swizzle(__float_as_int(x), (16 << 10) | 0x1F);
  return x + __int_as_float(y);
}

// Load one 16-element bf16 operand for a lane: two contiguous 8xbf16 runs
// at cbase and cbase+16 (matches the 16-bit A/B WMMA VGPR layout).
__device__ __forceinline__ v16bf load_op(const bf16* rowp, int cbase) {
  v8bf lo = *(const v8bf*)(rowp + cbase);
  v8bf hi = *(const v8bf*)(rowp + cbase + 16);
  return __builtin_shufflevector(lo, hi, 0, 1, 2, 3, 4, 5, 6, 7,
                                 8, 9, 10, 11, 12, 13, 14, 15);
}

__global__ void __launch_bounds__(256, 1)
satt_kernel(const float* __restrict__ cur,    // (B,H,P)
            const float* __restrict__ nf,     // (B,H,P,K)
            const float* __restrict__ vld,    // (B,1,P,K)
            const float* __restrict__ edg,    // (B,E,P,K)
            const float* __restrict__ w1,     // (64, 516)
            const float* __restrict__ b1,     // (64)
            const float* __restrict__ w2,     // (64)
            const float* __restrict__ b2p,    // (1)
            float* __restrict__ out)          // (B,H,P)
{
  extern __shared__ char smem[];
  bf16*  A1   = (bf16*)(smem + OFF_A1);
  bf16*  A2   = (bf16*)(smem + OFF_A2);
  bf16*  curT = (bf16*)(smem + OFF_CURT);
  float* curb = (float*)(smem + OFF_CURB);
  float* wts  = (float*)(smem + OFF_WT);
  float* pool = (float*)(smem + OFF_POOL);

  const int tid  = threadIdx.x;
  const int L    = tid & 31;         // lane
  const int w    = tid >> 5;         // wave (0..7)
  const int hi   = (L >> 4) & 1;     // lane half
  const int l15  = L & 15;
  const int bIdx = blockIdx.x >> 5;  // 32 p-tiles per batch
  const int p0   = (blockIdx.x & 31) * 32;

  bf16* NFb = (bf16*)(smem + OFF_NF + (size_t)w * 32 * ROWB);

  // ---- zero-fill both A buffers (covers the zero-padded edge chunk) ----
  {
    v8bf z;
#pragma unroll
    for (int i = 0; i < 8; ++i) z[i] = (bf16)0.0f;
    v8bf* dst = (v8bf*)(smem + OFF_A1);
    const int n = (2 * 64 * PITCH) / 8;
    for (int i = tid; i < n; i += 256) dst[i] = z;
  }
  __syncthreads();

  // ---- scatter w1 -> A2 (w1c), A1 (w1n at c<256, w1e at 256..259) ----
  for (int idx = tid; idx < 64 * 516; idx += 256) {
    int o = idx / 516;
    int c = idx - o * 516;
    bf16 h = (bf16)w1[idx];
    if (c < 256)       A2[o * PITCH + c] = h;
    else if (c < 512)  A1[o * PITCH + (c - 256)] = h;
    else               A1[o * PITCH + (256 + (c - 512))] = h;
  }

  // ---- stage current tile transposed: curT[p-local][c] bf16 ----
  {
    const float* cp = cur + (size_t)bIdx * 256 * PP1 + p0 + L;  // c-stride = P
    bf16* row = curT + L * PITCH;
#pragma unroll
    for (int i = 0; i < 4; ++i) {
      int c0 = w * 32 + i * 8;
      v8bf h;
#pragma unroll
      for (int j = 0; j < 8; ++j) h[j] = (bf16)cp[(size_t)(c0 + j) * PP1];
      *(v8bf*)(row + c0) = h;
    }
  }
  __syncthreads();

  // ---- small WMMA GEMM: curb[o][p-local] = (w1c @ current)[o][p] + b1[o] ----
  {
    const int ot = w & 3, ptl = w >> 2;
    v8f ca;
#pragma unroll
    for (int j = 0; j < 8; ++j) ca[j] = 0.0f;
    const bf16* arow = A2 + (ot * 16 + l15) * PITCH;
    const bf16* brow = curT + (ptl * 16 + l15) * PITCH;
#pragma unroll
    for (int ch = 0; ch < 8; ++ch) {
      int cbase = ch * 32 + hi * 8;
      v16bf a = load_op(arow, cbase);
      v16bf b = load_op(brow, cbase);
      ca = wmma_bf16(a, b, ca);
    }
#pragma unroll
    for (int j = 0; j < 8; ++j) {
      int o = ot * 16 + j + 8 * hi;
      curb[o * 33 + (ptl * 16 + l15)] = ca[j] + b1[o];
    }
  }
  __syncthreads();

  // ---- preload w2 (register-resident, ordered to match the acc layout) ----
  float w2r[32];
#pragma unroll
  for (int ot = 0; ot < 4; ++ot)
#pragma unroll
    for (int j = 0; j < 8; ++j) w2r[ot * 8 + j] = w2[ot * 16 + j + 8 * hi];
  const float b2v = b2p[0];

  // ---- 4 rounds: each wave processes one p-site per round ----
  for (int r = 0; r < 4; ++r) {
    const int s = r * 8 + w;   // p-local
    const int p = p0 + s;

    // Stage NF^T as bf16: row k = L, cols c (coalesced 128B global loads).
    {
      const float* nfp = nf + ((size_t)bIdx * 256 * PP1 + p) * 32 + L;
      bf16* row = NFb + L * PITCH;
      for (int c0 = 0; c0 < 256; c0 += 8) {
        v8bf h;
#pragma unroll
        for (int j = 0; j < 8; ++j) h[j] = (bf16)nfp[(size_t)(c0 + j) * (PP1 * 32)];
        *(v8bf*)(row + c0) = h;
      }
      // Append edge features as chunk-8 rows (c=256..259), zero pad to 287.
      v8bf he;
#pragma unroll
      for (int c = 0; c < 4; ++c)
        he[c] = (bf16)edg[(((size_t)bIdx * 4 + c) * PP1 + p) * 32 + L];
#pragma unroll
      for (int c = 4; c < 8; ++c) he[c] = (bf16)0.0f;
      v8bf z;
#pragma unroll
      for (int i = 0; i < 8; ++i) z[i] = (bf16)0.0f;
      *(v8bf*)(row + 256) = he;
      *(v8bf*)(row + 264) = z;
      *(v8bf*)(row + 272) = z;
      *(v8bf*)(row + 280) = z;
    }

    // Main GEMM: hid(64 x 32) = [w1n|w1e] @ [NF;edge], 9 K-chunks of 32.
    v8f acc[4][2];
#pragma unroll
    for (int ot = 0; ot < 4; ++ot)
#pragma unroll
      for (int kt = 0; kt < 2; ++kt)
#pragma unroll
        for (int j = 0; j < 8; ++j) acc[ot][kt][j] = 0.0f;

    for (int ch = 0; ch < 9; ++ch) {
      int cbase = ch * 32 + hi * 8;
      v16bf bop0 = load_op(NFb + (0 * 16 + l15) * PITCH, cbase);
      v16bf bop1 = load_op(NFb + (1 * 16 + l15) * PITCH, cbase);
#pragma unroll
      for (int ot = 0; ot < 4; ++ot) {
        v16bf aop = load_op(A1 + (ot * 16 + l15) * PITCH, cbase);
        acc[ot][0] = wmma_bf16(aop, bop0, acc[ot][0]);
        acc[ot][1] = wmma_bf16(aop, bop1, acc[ot][1]);
      }
    }

    // Epilogue: + cur/b1, ReLU, contract over o with w2.
    float lg0 = 0.0f, lg1 = 0.0f;
#pragma unroll
    for (int ot = 0; ot < 4; ++ot) {
#pragma unroll
      for (int j = 0; j < 8; ++j) {
        float cb = curb[(ot * 16 + j + 8 * hi) * 33 + s];
        float wv = w2r[ot * 8 + j];
        float h0 = fmaxf(acc[ot][0][j] + cb, 0.0f);
        float h1 = fmaxf(acc[ot][1][j] + cb, 0.0f);
        lg0 += h0 * wv;
        lg1 += h1 * wv;
      }
    }
    // Combine lane halves (each half holds 32 of the 64 o-terms for k=l15+16*kt).
    lg0 = xor16_pair_add(lg0);
    lg1 = xor16_pair_add(lg1);
    float lgt = (hi == 0 ? lg0 : lg1) + b2v;  // this lane's k == L

    // Masked softmax over K = wave (32 lanes).
    float vv = vld[((size_t)bIdx * PP1 + p) * 32 + L];
    bool isv = vv > 0.5f;
    unsigned long long bal = __ballot(isv);
    bool any = (bal != 0ull);
    float ls = any ? (isv ? lgt : -__builtin_inff()) : 0.0f;
    float mx = ls;
    mx = xor_max<1>(mx); mx = xor_max<2>(mx); mx = xor_max<4>(mx);
    mx = xor_max<8>(mx); mx = xor_max<16>(mx);
    float e = __expf(ls - mx);
    float sm = e;
    sm = xor_add<1>(sm); sm = xor_add<2>(sm); sm = xor_add<4>(sm);
    sm = xor_add<8>(sm); sm = xor_add<16>(sm);
    float wt = (any && isv) ? (e / sm) : 0.0f;
    wts[w * 32 + L] = wt;

    // Pooling from the LDS bf16 tile: lane owns c = 8L..8L+7.
    float pc[8];
#pragma unroll
    for (int i = 0; i < 8; ++i) pc[i] = 0.0f;
    for (int k = 0; k < 32; ++k) {
      float wk = wts[w * 32 + k];
      v8bf nb = *(const v8bf*)(NFb + k * PITCH + L * 8);
#pragma unroll
      for (int i = 0; i < 8; ++i) pc[i] += (float)nb[i] * wk;
    }
#pragma unroll
    for (int i = 0; i < 8; ++i) pool[(L * 8 + i) * 33 + s] = pc[i];
  }
  __syncthreads();

  // ---- coalesced writeout: one 128B row (32 consecutive p) per instruction ----
  for (int i = 0; i < 32; ++i) {
    int c = w * 32 + i;
    out[((size_t)bIdx * 256 + c) * PP1 + p0 + L] = pool[c * 33 + L];
  }
}

extern "C" void kernel_launch(void* const* d_in, const int* in_sizes, int n_in,
                              void* d_out, int out_size, void* d_ws, size_t ws_size,
                              hipStream_t stream) {
  const float* cur = (const float*)d_in[0];
  const float* nf  = (const float*)d_in[1];
  const float* vld = (const float*)d_in[2];
  const float* edg = (const float*)d_in[3];
  const float* w1  = (const float*)d_in[4];
  const float* b1  = (const float*)d_in[5];
  const float* w2  = (const float*)d_in[6];
  const float* b2  = (const float*)d_in[7];
  float* out = (float*)d_out;

  (void)in_sizes; (void)n_in; (void)out_size; (void)d_ws; (void)ws_size;

  hipFuncSetAttribute((const void*)satt_kernel,
                      hipFuncAttributeMaxDynamicSharedMemorySize, SMEM_TOTAL);
  // Grid: B * (P / 32) = 8 * 32 = 256 workgroups of 8 waves.
  satt_kernel<<<dim3(BB * (PP1 / 32)), dim3(256), SMEM_TOTAL, stream>>>(
      cur, nf, vld, edg, w1, b1, w2, b2, out);
}
